// SICE_1769526526741
// MI455X (gfx1250) — matrix operand order repaired and checked by reading, not compiled, not run on hardware
//
#include <hip/hip_runtime.h>

// ---------------------------------------------------------------------------
// Types / helpers
// ---------------------------------------------------------------------------
typedef __attribute__((ext_vector_type(2)))  float  v2f;
typedef __attribute__((ext_vector_type(8)))  float  v8f;
typedef __attribute__((ext_vector_type(16))) __bf16 v16bf;
typedef __attribute__((ext_vector_type(4)))  unsigned int v4u;
typedef __attribute__((ext_vector_type(4)))  int    v4i;
typedef __attribute__((ext_vector_type(8)))  int    v8i;

union BFx16 { v16bf v; uint4 q[2]; };

__device__ __forceinline__ unsigned short f2bf(float f) {
  union { float f; unsigned u; } c; c.f = f;
  unsigned u = c.u;
  unsigned r = u + 0x7fffu + ((u >> 16) & 1u);   // round-nearest-even
  return (unsigned short)(r >> 16);
}

__device__ __forceinline__ float blockReduceSum256(float v, float* sm) {
  const int tid = threadIdx.x;
  sm[tid] = v; __syncthreads();
  for (int s = 128; s > 0; s >>= 1) {
    if (tid < s) sm[tid] += sm[tid + s];
    __syncthreads();
  }
  return sm[0];
}

// LDS byte offset of a __shared__ object (generic -> addrspace(3) strips aperture)
__device__ __forceinline__ unsigned lds_addr_of(const void* p) {
  return (unsigned)(unsigned long long)(const __attribute__((address_space(3))) void*)p;
}

// ---------------------------------------------------------------------------
// Tensor Data Mover: async 2D tile load Global->LDS with hardware row padding.
// Descriptor per CDNA5 ISA ch.8 (groups 0/1; groups 2/3 zero for 2D tensors).
//   tile_d0: contiguous elements per row, tile_d1: rows
//   row_stride: elements between rows in memory
//   ds_log2: log2(element bytes); pad_int/pad_amt: D# pad codes (DWORD units)
// ---------------------------------------------------------------------------
__device__ __forceinline__ void tdm_load_2d(unsigned lds_byte_addr, const void* gptr,
                                            unsigned tile_d0, unsigned tile_d1,
                                            unsigned long long row_stride,
                                            unsigned ds_log2,
                                            unsigned pad_int, unsigned pad_amt) {
  const unsigned long long ga = (unsigned long long)gptr;
  v4u g0;
  g0.x = 1u;                                                  // count=1 (valid user D#)
  g0.y = lds_byte_addr;                                       // lds_addr
  g0.z = (unsigned)(ga & 0xffffffffu);                        // global_addr[31:0]
  g0.w = (unsigned)((ga >> 32) & 0x01ffffffu) | (2u << 30);   // addr[56:32] | type=2
  const unsigned long long st = row_stride << ds_log2 >> ds_log2; // keep as elements
  const unsigned td0 = 0x7fffffffu, td1 = 0x7fffffffu;        // huge => no OOB clip
  v8i g1;
  g1[0] = (int)((ds_log2 << 16) | (1u << 20) | (pad_int << 22) | (pad_amt << 25));
  g1[1] = (int)((td0 & 0xffffu) << 16);                       // barrier_addr=0 | td0.lo
  g1[2] = (int)(((td0 >> 16) & 0xffffu) | ((td1 & 0xffffu) << 16));
  g1[3] = (int)(((td1 >> 16) & 0xffffu) | (tile_d0 << 16));   // td1.hi | tile_dim0
  g1[4] = (int)(tile_d1 & 0xffffu);                           // tile_dim1 | tile_dim2=0
  g1[5] = (int)((unsigned long long)st & 0xffffffffu);        // dim0_stride[31:0]
  g1[6] = (int)(((unsigned long long)st >> 32) & 0xffffu);    // dim0_stride[47:32]
  g1[7] = 0;
  v4i g2 = {0, 0, 0, 0};
  v4i g3 = {0, 0, 0, 0};
#if defined(__clang_major__) && (__clang_major__ >= 23)
  v8i g4 = {0, 0, 0, 0, 0, 0, 0, 0};
  __builtin_amdgcn_tensor_load_to_lds(g0, g1, g2, g3, g4, 0);
#else
  __builtin_amdgcn_tensor_load_to_lds(g0, g1, g2, g3, 0);
#endif
}

// ---------------------------------------------------------------------------
// Conv 1x1 as GEMM:  y[b,d,hw] = sum_c w[d,c] * x[b,c,hw]
// M=256(d) K=2048(c), per-batch N=784(hw).  bf16 WMMA 16x16x32, fp32 accum.
// Weights staged by double-buffered TDM; x staged cooperatively (needs f32->bf16).
// ---------------------------------------------------------------------------
__global__ __launch_bounds__(256)
void conv_gemm_bf16(float* __restrict__ y,
                    const unsigned short* __restrict__ wbf,
                    const float* __restrict__ x)
{
  __shared__ unsigned short Asb[2][128 * 40];   // [row][k], pad 32->40 bf16 (TDM pad)
  __shared__ unsigned short Bsb[112 * 40];      // [n][k] transposed, padded

  const int tid = threadIdx.x;
  const int w   = tid >> 5;
  const int l   = tid & 31;
  const int ln  = l & 15;
  const int h   = l >> 4;
  const int m0  = blockIdx.x * 128;
  const int hw0 = blockIdx.y * 112;
  const int b   = blockIdx.z;

  const unsigned asBase = lds_addr_of(&Asb[0][0]);
  const unsigned asBufB = 128 * 40 * 2;                 // bytes per A buffer

  v8f acc[7];
#pragma unroll
  for (int t = 0; t < 7; ++t) acc[t] = {};

  const int bkk = tid >> 3, bnseg = tid & 7;            // B stage: 32 k x 8 segs of 14

  // prefetch A slice 0 via TDM: tile 32(bf16) x 128 rows, stride 2048, pad 16->20 DW
  if (w == 0)
    tdm_load_2d(asBase, wbf + (size_t)m0 * 2048, 32, 128, 2048ull, 1, 3, 3);

  for (int i = 0; i < 64; ++i) {
    const int k0 = i * 32;
    { // stage x fp32 -> bf16, transposed to [n][k]
      const float* g = x + ((size_t)b * 2048 + (size_t)(k0 + bkk)) * 784 + hw0 + bnseg * 14;
#pragma unroll
      for (int j = 0; j < 14; ++j)
        Bsb[(bnseg * 14 + j) * 40 + bkk] = f2bf(g[j]);
    }
    if (w == 0) {
      if (i + 1 < 64) {
        tdm_load_2d(asBase + ((i + 1) & 1) * asBufB,
                    wbf + (size_t)m0 * 2048 + (k0 + 32), 32, 128, 2048ull, 1, 3, 3);
        __builtin_amdgcn_s_wait_tensorcnt(1);           // slice i retired (in-order)
      } else {
        __builtin_amdgcn_s_wait_tensorcnt(0);
      }
    }
    __syncthreads();                                    // A[i] + B[i] ready

    const unsigned short* asb = &Asb[i & 1][0];
    BFx16 af;
    af.q[0] = *(const uint4*)&asb[(w * 16 + ln) * 40 + h * 8];
    af.q[1] = *(const uint4*)&asb[(w * 16 + ln) * 40 + 16 + h * 8];
#pragma unroll
    for (int t = 0; t < 7; ++t) {
      const int n = t * 16 + ln;
      BFx16 bf;
      bf.q[0] = *(const uint4*)&Bsb[n * 40 + h * 8];
      bf.q[1] = *(const uint4*)&Bsb[n * 40 + 16 + h * 8];
      acc[t] = __builtin_amdgcn_wmma_f32_16x16x32_bf16(
          false, af.v, false, bf.v, (short)0, acc[t], false, false);
    }
    __syncthreads();                                    // done reading B buffer
  }

#pragma unroll
  for (int t = 0; t < 7; ++t) {
#pragma unroll
    for (int v = 0; v < 8; ++v) {
      const int d  = m0 + w * 16 + v + h * 8;
      const int hw = hw0 + t * 16 + ln;
      y[((size_t)b * 256 + d) * 784 + hw] = acc[t][v];
    }
  }
}

// ---------------------------------------------------------------------------
// Generic batched 256x256 fp32 WMMA matmul, fully TDM-fed & double-buffered:
//   E[b] = alpha * A[b] @ op(B[b]) + diag_add * I + s_scale * S[b]
// TRANSB=1: B given NxK row-major (gram) -> LDS [n][k] stride 36
// TRANSB=0: B given KxN row-major        -> LDS [k][n] stride 260 (no transpose!)
// Grid (4, NB): each WG computes a 64x256 strip. K multiple of 32.
// ---------------------------------------------------------------------------
template <int TRANSB>
__global__ __launch_bounds__(256)
void bmm256(float* __restrict__ E,
            const float* __restrict__ A,
            const float* __restrict__ Bm,
            int K, int lda, int ldb,
            unsigned long long sA, unsigned long long sB,
            float alpha, float diag_add,
            const float* __restrict__ S, float s_scale)
{
  __shared__ float As[2][64 * 36];    // [row][k], pad 32->36 DWORDs (TDM pad)
  __shared__ float Bs[2][256 * 36];   // TRANSB: [n][k]/36 ; else [k][n]/260 (8320<9216)

  const int tid = threadIdx.x;
  const int w = tid >> 5, l = tid & 31, ln = l & 15, h = l >> 4;
  const int m0 = blockIdx.x * 64;
  const int b  = blockIdx.y;
  const float* Ab = A  + (size_t)b * sA;
  const float* Bb = Bm + (size_t)b * sB;

  const unsigned asBase = lds_addr_of(&As[0][0]);
  const unsigned bsBase = lds_addr_of(&Bs[0][0]);
  const unsigned asBufB = 64 * 36 * 4, bsBufB = 256 * 36 * 4;

  v8f acc[8];
#pragma unroll
  for (int t = 0; t < 8; ++t) acc[t] = {};

  const int nbase = (w & 1) * 128;
  const int mloc  = (w >> 1) * 16 + ln;
  const int nSl   = K >> 5;

  auto issue = [&](int slice, int buf) {
    const int k0 = slice * 32;
    // A: 64 rows x 32 f32, stride lda, pad 32->36 DW (interval code 4, amount 3)
    tdm_load_2d(asBase + buf * asBufB, Ab + (size_t)m0 * lda + k0,
                32, 64, (unsigned long long)lda, 2, 4, 3);
    if (TRANSB) // B^T rows: 256 rows x 32 f32, stride ldb, same padding
      tdm_load_2d(bsBase + buf * bsBufB, Bb + k0,
                  32, 256, (unsigned long long)ldb, 2, 4, 3);
    else        // B rows: 32 rows x 256 f32, stride ldb, pad 256->260 DW (code 7)
      tdm_load_2d(bsBase + buf * bsBufB, Bb + (size_t)k0 * ldb,
                  256, 32, (unsigned long long)ldb, 2, 7, 3);
  };

  if (w == 0) issue(0, 0);

  for (int i = 0; i < nSl; ++i) {
    if (w == 0) {
      if (i + 1 < nSl) {
        issue(i + 1, (i + 1) & 1);
        __builtin_amdgcn_s_wait_tensorcnt(2);   // slice i's 2 TDM ops retired
      } else {
        __builtin_amdgcn_s_wait_tensorcnt(0);
      }
    }
    __syncthreads();                            // buffers for slice i ready

    const float* as = &As[i & 1][0];
    const float* bs = &Bs[i & 1][0];
#pragma unroll
    for (int ks = 0; ks < 8; ++ks) {
      const int kk = ks * 4;
      v2f av = *(const v2f*)&as[mloc * 36 + kk + h * 2];
#pragma unroll
      for (int t = 0; t < 8; ++t) {
        const int n = nbase + t * 16 + ln;
        v2f bv;
        if (TRANSB) {
          bv = *(const v2f*)&bs[n * 36 + kk + h * 2];
        } else {
          bv.x = bs[(kk + h * 2) * 260 + n];
          bv.y = bs[(kk + h * 2 + 1) * 260 + n];
        }
        acc[t] = __builtin_amdgcn_wmma_f32_16x16x4_f32(
            false, av, false, bv, (short)0, acc[t], false, false);
      }
    }
    __syncthreads();                            // done reading slice i buffers
  }

#pragma unroll
  for (int t = 0; t < 8; ++t) {
#pragma unroll
    for (int v = 0; v < 8; ++v) {
      const int row = m0 + (w >> 1) * 16 + v + h * 8;
      const int col = nbase + t * 16 + ln;
      float val = alpha * acc[t][v];
      if (row == col) val += diag_add;
      if (S) val += s_scale * S[(size_t)b * 65536 + (size_t)row * 256 + col];
      E[(size_t)b * 65536 + (size_t)row * 256 + col] = val;
    }
  }
}

// ---------------------------------------------------------------------------
// Small kernels
// ---------------------------------------------------------------------------
__global__ void w_to_bf16(unsigned short* __restrict__ o, const float* __restrict__ w) {
  const int i = blockIdx.x * 256 + threadIdx.x;
  o[i] = f2bf(w[i]);
}

__global__ void bn_stats(const float* __restrict__ y, float* __restrict__ mu,
                         float* __restrict__ rsig) {
  __shared__ float sm[256];
  const int d = blockIdx.x;
  float s = 0.f, s2 = 0.f;
  for (int idx = threadIdx.x; idx < 32 * 784; idx += 256) {
    const int bb = idx / 784, hw = idx - bb * 784;
    const float v = y[((size_t)bb * 256 + d) * 784 + hw];
    s += v; s2 += v * v;
  }
  const float S1 = blockReduceSum256(s, sm);
  const float S2 = blockReduceSum256(s2, sm);
  if (threadIdx.x == 0) {
    const float m = S1 / 25088.f;
    const float var = S2 / 25088.f - m * m;
    mu[d] = m;
    rsig[d] = rsqrtf(var + 1e-5f);
  }
}

__global__ void bn_apply_center(const float* __restrict__ y, float* __restrict__ z,
                                const float* __restrict__ gamma, const float* __restrict__ beta,
                                const float* __restrict__ mu, const float* __restrict__ rsig,
                                float* __restrict__ rmu) {
  __shared__ float sm[256];
  const int d = blockIdx.x, b = blockIdx.y;
  const float g = gamma[d], be = beta[d], m = mu[d], rs = rsig[d];
  float s = 0.f;
  for (int hw = threadIdx.x; hw < 800; hw += 256) {
    float zv = 0.f;
    if (hw < 784) {
      const float v = y[((size_t)b * 256 + d) * 784 + hw];
      zv = fmaxf(g * (v - m) * rs + be, 0.f);
      s += zv;
    }
    z[((size_t)b * 256 + d) * 800 + hw] = zv;
  }
  const float S1 = blockReduceSum256(s, sm);
  if (threadIdx.x == 0) rmu[b * 256 + d] = S1 / 784.f;
}

__global__ void trace256(const float* __restrict__ X, float* __restrict__ t) {
  __shared__ float sm[256];
  const int b = blockIdx.x;
  const float v = X[(size_t)b * 65536 + (size_t)threadIdx.x * 257];
  const float s = blockReduceSum256(v, sm);
  if (threadIdx.x == 0) t[b] = s;
}

// mode 0: *s   mode 1: *1/s   mode 2: *rsqrt(s)
__global__ void scale_mat(float* __restrict__ E, const float* __restrict__ X,
                          const float* __restrict__ t, int mode) {
  const int b = blockIdx.y;
  const size_t e = (size_t)b * 65536 + blockIdx.x * 256 + threadIdx.x;
  const float s = t[b];
  const float f = (mode == 0) ? s : (mode == 1) ? (1.0f / s) : rsqrtf(s);
  E[e] = X[e] * f;
}

__global__ void axpy_diag(float* __restrict__ E, const float* __restrict__ X,
                          float a, float d) {
  const int b = blockIdx.y;
  const int idx = blockIdx.x * 256 + threadIdx.x;
  const int i = idx >> 8, j = idx & 255;
  const size_t e = (size_t)b * 65536 + idx;
  E[e] = a * X[e] + ((i == j) ? d : 0.0f);
}

__global__ void copy_add_jitter(float* __restrict__ E, const float* __restrict__ X,
                                const float* __restrict__ jit) {
  const int b = blockIdx.y;
  const int idx = blockIdx.x * 256 + threadIdx.x;
  const int i = idx >> 8, j = idx & 255;
  const size_t e = (size_t)b * 65536 + idx;
  E[e] = X[e] + ((i == j) ? (1e-10f + 1e-9f * jit[i]) : 0.0f);
}

__global__ void finish_cov(float* __restrict__ C, const float* __restrict__ rmu) {
  const int b = blockIdx.y;
  const int idx = blockIdx.x * 256 + threadIdx.x;
  const int i = idx >> 8, j = idx & 255;
  C[(size_t)b * 65536 + idx] -= rmu[b * 256 + i] * rmu[b * 256 + j];
}

__global__ void prox_sym(float* __restrict__ O, const float* __restrict__ L,
                         const float* __restrict__ G, float dec, float sp) {
  const int b = blockIdx.y;
  const int idx = blockIdx.x * 256 + threadIdx.x;
  const int i = idx >> 8, j = idx & 255;
  const size_t base = (size_t)b * 65536;
  const float lij = L[base + idx],        lji = L[base + j * 256 + i];
  const float gij = G[base + idx],        gji = G[base + j * 256 + i];
  const float f1 = fmaxf(fmaxf(lij, 0.f) - dec * (gij + sp), 0.f)
                 - fmaxf(fmaxf(-lij, 0.f) - dec * (-gij + sp), 0.f);
  const float f2 = fmaxf(fmaxf(lji, 0.f) - dec * (gji + sp), 0.f)
                 - fmaxf(fmaxf(-lji, 0.f) - dec * (-gji + sp), 0.f);
  O[base + idx] = 0.5f * (f1 + f2);
}

__global__ void triu_out(float* __restrict__ o, const float* __restrict__ L,
                         const float* __restrict__ t) {
  const int b = blockIdx.y;
  const int idx = blockIdx.x * 256 + threadIdx.x;
  const int i = idx >> 8, j = idx & 255;
  if (j >= i) {
    const int k = i * 256 - (i * (i - 1)) / 2 + (j - i);
    o[(size_t)b * 32896 + k] = L[(size_t)b * 65536 + idx] * rsqrtf(t[b]);
  }
}

// ---------------------------------------------------------------------------
// Host orchestration
// ---------------------------------------------------------------------------
extern "C" void kernel_launch(void* const* d_in, const int* in_sizes, int n_in,
                              void* d_out, int out_size, void* d_ws, size_t ws_size,
                              hipStream_t stream) {
  (void)in_sizes; (void)n_in; (void)out_size; (void)ws_size;
  const float* x      = (const float*)d_in[0];
  const float* conv_w = (const float*)d_in[1];
  const float* gamma  = (const float*)d_in[2];
  const float* beta   = (const float*)d_in[3];
  const float* jitter = (const float*)d_in[4];
  float* out = (float*)d_out;

  char* p = (char*)d_ws;
  auto alloc = [&](size_t bytes) -> char* {
    char* r = p; p += (bytes + 255) & ~(size_t)255; return r;
  };
  const size_t MAT = (size_t)32 * 65536 * sizeof(float);
  unsigned short* wbf = (unsigned short*)alloc((size_t)256 * 2048 * 2);
  float* y    = (float*)alloc((size_t)32 * 256 * 784 * 4);
  float* z    = (float*)alloc((size_t)32 * 256 * 800 * 4);
  float* mu   = (float*)alloc(256 * 4);
  float* rsig = (float*)alloc(256 * 4);
  float* rmu  = (float*)alloc(32 * 256 * 4);
  float* ts0  = (float*)alloc(32 * 4);
  float* ts1  = (float*)alloc(32 * 4);
  float* C    = (float*)alloc(MAT);
  float* Xj   = (float*)alloc(MAT);
  float* Am   = (float*)alloc(MAT);
  float* Y0   = (float*)alloc(MAT);
  float* Y1   = (float*)alloc(MAT);
  float* Z0   = (float*)alloc(MAT);
  float* Z1   = (float*)alloc(MAT);
  float* Tm   = (float*)alloc(MAT);
  float* zz   = (float*)alloc(MAT);
  float* LLT  = (float*)alloc(MAT);
  float* LLT2 = (float*)alloc(MAT);
  float* G12  = (float*)alloc(MAT);

  const dim3 ew(256, 32);

  auto sq = [&](float* E, const float* A, const float* B,
                float alpha, float diag, const float* S, float ss) {
    bmm256<0><<<dim3(4, 32), 256, 0, stream>>>(E, A, B, 256, 256, 256,
                                               65536ull, 65536ull, alpha, diag, S, ss);
  };
  auto swapf = [](float*& a, float*& b) { float* t = a; a = b; b = t; };

  auto invsqrtm = [&](const float* X, float* ozz) {
    trace256<<<32, 256, 0, stream>>>(X, ts1);
    scale_mat<<<ew, 256, 0, stream>>>(Am, X, ts1, 1);        // A = X / tr
    axpy_diag<<<ew, 256, 0, stream>>>(Tm, Am, -0.5f, 1.5f);  // ZY = 0.5(3I - A)
    sq(Y0, Am, Tm, 1.f, 0.f, nullptr, 0.f);                  // Y = A @ ZY
    axpy_diag<<<ew, 256, 0, stream>>>(Z0, Tm, 1.0f, 0.0f);   // Z = ZY
    float *Yc = Y0, *Yn = Y1, *Zc = Z0, *Zn = Z1;
    for (int it = 0; it < 5; ++it) {
      sq(Tm, Zc, Yc, -0.5f, 1.5f, nullptr, 0.f);             // ZY = 0.5(3I - Z@Y)
      sq(Yn, Yc, Tm, 1.f, 0.f, nullptr, 0.f);                // Y' = Y @ ZY
      sq(Zn, Tm, Zc, 1.f, 0.f, nullptr, 0.f);                // Z' = ZY @ Z
      swapf(Yc, Yn); swapf(Zc, Zn);
    }
    sq(Tm, Zc, Yc, -0.5f, 1.5f, nullptr, 0.f);
    sq(ozz, Tm, Zc, 1.f, 0.f, nullptr, 0.f);
    scale_mat<<<ew, 256, 0, stream>>>(ozz, ozz, ts1, 2);     // * tr^-0.5
  };

  // ---- 1) conv 1x1 (bf16 WMMA, TDM-fed weights) + BN + ReLU ----
  w_to_bf16<<<2048, 256, 0, stream>>>(wbf, conv_w);
  conv_gemm_bf16<<<dim3(2, 7, 32), 256, 0, stream>>>(y, wbf, x);
  bn_stats<<<256, 256, 0, stream>>>(y, mu, rsig);
  bn_apply_center<<<dim3(256, 32), 256, 0, stream>>>(y, z, gamma, beta, mu, rsig, rmu);

  // ---- 2) covpool: C = Z Z^T / 784 - mu mu^T ; C /= trace(C) ----
  bmm256<1><<<dim3(4, 32), 256, 0, stream>>>(C, z, z, 800, 800, 800,
                                             204800ull, 204800ull,
                                             1.0f / 784.0f, 0.f, nullptr, 0.f);
  finish_cov<<<ew, 256, 0, stream>>>(C, rmu);
  trace256<<<32, 256, 0, stream>>>(C, ts0);
  scale_mat<<<ew, 256, 0, stream>>>(C, C, ts0, 1);

  // ---- 3) SICE ----
  copy_add_jitter<<<ew, 256, 0, stream>>>(Xj, C, jitter);
  invsqrtm(Xj, zz);
  sq(LLT, zz, zz, 1.f, 0.f, nullptr, 0.f);
  float *Lc = LLT, *Ln = LLT2;
  for (int i = 0; i < 3; ++i) {
    copy_add_jitter<<<ew, 256, 0, stream>>>(Xj, Lc, jitter);
    invsqrtm(Xj, zz);
    sq(G12, zz, zz, -1.f, 0.f, C, 1.f);
    const float dec = 5.0f * (1.0f - (float)i / 2.0f);
    prox_sym<<<ew, 256, 0, stream>>>(Ln, Lc, G12, dec, 0.07f);
    swapf(Lc, Ln);
  }

  // ---- 4) trace normalize + triuvec ----
  trace256<<<32, 256, 0, stream>>>(Lc, ts0);
  triu_out<<<ew, 256, 0, stream>>>(out, Lc, ts0);
}
